// SelfAttention_3152505996010
// MI455X (gfx1250) — matrix operand order repaired and checked by reading, not compiled, not run on hardware
//
#include <hip/hip_runtime.h>

#define BB 4
#define SS 2048
#define CC 256
#define HH 8
#define HC (HH*CC)   // 2048
#define MM (BB*SS)   // 8192
#define KCHUNK 256

typedef __attribute__((ext_vector_type(16))) __bf16 v16bf;
typedef __attribute__((ext_vector_type(8)))  float  v8f;
typedef unsigned short us;

union U16 { v16bf v; uint4 q[2]; };

__device__ inline us f2bf(float f){
  unsigned u = __float_as_uint(f);
  u += 0x7FFFu + ((u >> 16) & 1u);
  return (us)(u >> 16);
}

// build a v16bf A/B operand from two 16-byte chunks
__device__ inline v16bf load2(const us* p0, const us* p1){
  U16 u;
  u.q[0] = *(const uint4*)p0;
  u.q[1] = *(const uint4*)p1;
  return u.v;
}

__device__ inline v8f bfwmma(v16bf a, v16bf b, v8f c){
  return __builtin_amdgcn_wmma_f32_16x16x32_bf16(false, a, false, b, (short)0, c, false, false);
}

// ---------------- stage 0: conversions ----------------
__global__ void cvt_x(const float* __restrict__ x, us* __restrict__ xbf, int n){
  int i = blockIdx.x*blockDim.x + threadIdx.x;
  if (i < n) xbf[i] = f2bf(x[i]);
}

// src: (K x N) row-major f32 ; dst: (N x K) row-major bf16, scaled
__global__ void cvt_wT(const float* __restrict__ w, us* __restrict__ wt,
                       int K, int N, float scale){
  int i = blockIdx.x*blockDim.x + threadIdx.x;
  if (i < K*N){
    int n = i / K, k = i % K;
    wt[i] = f2bf(w[(size_t)k*N + n] * scale);
  }
}

// ---------------- stage 1: QKV projection GEMM ----------------
// one wave -> one 16x64 tile (A operand reused across 4 B tiles)
__global__ __launch_bounds__(32) void qkv_gemm(
    const us* __restrict__ xbf,
    const us* __restrict__ Wqt, const us* __restrict__ Wkt, const us* __restrict__ Wvt,
    const float* __restrict__ bq, const float* __restrict__ bk, const float* __restrict__ bv,
    us* __restrict__ qO, us* __restrict__ kO, us* __restrict__ vtO)
{
  int lane = threadIdx.x;
  int hf = lane >> 4, l16 = lane & 15;
  int nBase = blockIdx.x * 64;
  int mBase = blockIdx.y * 16;
  int z = blockIdx.z;
  const us* Wt = (z==0) ? Wqt : (z==1) ? Wkt : Wvt;
  const float* bias = (z==0) ? bq : (z==1) ? bk : bv;
  float bscale = (z==2) ? 1.0f : 0.25f;   // q,k scale folded (256^-0.25)

  const us* arow = xbf + (size_t)(mBase + l16) * CC;
  v8f acc[4] = {};
  #pragma unroll
  for (int cs = 0; cs < CC/32; ++cs){
    v16bf a = load2(arow + cs*32 + hf*8, arow + cs*32 + 16 + hf*8);
    #pragma unroll
    for (int t = 0; t < 4; ++t){
      const us* bp = Wt + (size_t)(nBase + t*16 + l16) * CC + cs*32 + hf*8;
      acc[t] = bfwmma(a, load2(bp, bp + 16), acc[t]);
    }
  }
  #pragma unroll
  for (int t = 0; t < 4; ++t){
    int n = nBase + t*16 + l16;
    int h = n / CC, c = n % CC;
    float bb = bias[n] * bscale;
    #pragma unroll
    for (int r = 0; r < 8; ++r){
      int m = mBase + r + hf*8;
      int b = m / SS, s = m % SS;
      us val = f2bf(acc[t][r] + bb);
      if (z == 2)      vtO[(((size_t)b*HH + h)*CC + c)*SS + s] = val;  // v transposed
      else if (z == 1) kO [(((size_t)b*HH + h)*SS + s)*CC + c] = val;
      else             qO [(((size_t)b*HH + h)*SS + s)*CC + c] = val;
    }
  }
}

// ---------------- stage 2: flash attention ----------------
// block = 128 threads (4 waves) handles one (b,h, 16-query tile)
__global__ __launch_bounds__(128) void attn_kernel(
    const us* __restrict__ q, const us* __restrict__ kk,
    const us* __restrict__ vt, us* __restrict__ attn)
{
  __shared__ float sS[16*KCHUNK];
  __shared__ us    sP[16*KCHUNK];
  __shared__ float sM[16], sL[16], sA[16];
  __shared__ float sRed[8][16];

  int bid = blockIdx.x;
  int qt = bid & (SS/16 - 1);
  int bh = bid / (SS/16);
  int b = bh / HH, h = bh % HH;
  int tid = threadIdx.x;
  int lane = tid & 31, wave = tid >> 5;
  int hf = lane >> 4, l16 = lane & 15;
  int row = tid & 15, seg = tid >> 4;   // softmax: 16 rows x 8 column-segments

  const us* qbase = q  + (size_t)bh * SS * CC;
  const us* kbase = kk + (size_t)bh * SS * CC;
  const us* vbase = vt + (size_t)bh * CC * SS;

  // Q tile (16 rows x 256) resident in registers as 8 A-operands
  v16bf Qa[CC/32];
  const us* qrow = qbase + (size_t)(qt*16 + l16) * CC;
  #pragma unroll
  for (int cs = 0; cs < CC/32; ++cs)
    Qa[cs] = load2(qrow + cs*32 + hf*8, qrow + cs*32 + 16 + hf*8);

  v8f O[4] = {};   // this wave owns c-dims wave*64 .. wave*64+63

  if (tid < 16){ sM[tid] = -3.0e38f; sL[tid] = 0.f; }
  __syncthreads();

  for (int kc = 0; kc < SS/KCHUNK; ++kc){
    // --- scores: each wave computes 4 key blocks of 16, reusing Qa ---
    #pragma unroll
    for (int jb = 0; jb < 4; ++jb){
      int keyLocal = (wave*4 + jb) * 16;
      const us* krow = kbase + (size_t)(kc*KCHUNK + keyLocal + l16) * CC;
      v8f acc = {};
      #pragma unroll
      for (int cs = 0; cs < CC/32; ++cs){
        const us* bp = krow + cs*32 + hf*8;
        acc = bfwmma(Qa[cs], load2(bp, bp + 16), acc);
      }
      #pragma unroll
      for (int r = 0; r < 8; ++r)
        sS[(r + hf*8)*KCHUNK + keyLocal + l16] = acc[r];
    }
    __syncthreads();

    // --- online softmax over this chunk, parallel over 128 threads ---
    {
      float mc = -3.0e38f;
      for (int j = 0; j < 32; ++j)
        mc = fmaxf(mc, sS[row*KCHUNK + seg*32 + j]);
      sRed[seg][row] = mc;
    }
    __syncthreads();
    if (tid < 16){
      float mOld = sM[tid];
      float m = -3.0e38f;
      #pragma unroll
      for (int s2 = 0; s2 < 8; ++s2) m = fmaxf(m, sRed[s2][tid]);
      float mNew = fmaxf(mOld, m);
      sA[tid] = __expf(mOld - mNew);
      sM[tid] = mNew;
    }
    __syncthreads();
    {
      float mNew = sM[row];
      float sum = 0.f;
      for (int j = 0; j < 32; ++j){
        float p = __expf(sS[row*KCHUNK + seg*32 + j] - mNew);
        sum += p;
        sP[row*KCHUNK + seg*32 + j] = f2bf(p);
      }
      sRed[seg][row] = sum;
    }
    __syncthreads();
    if (tid < 16){
      float sum = 0.f;
      #pragma unroll
      for (int s2 = 0; s2 < 8; ++s2) sum += sRed[s2][tid];
      sL[tid] = sL[tid]*sA[tid] + sum;
    }
    __syncthreads();

    // --- rescale accumulators, then P @ V (A operand hoisted, reused 4x) ---
    #pragma unroll
    for (int nt = 0; nt < 4; ++nt)
      #pragma unroll
      for (int r = 0; r < 8; ++r) O[nt][r] *= sA[r + hf*8];

    #pragma unroll
    for (int ks = 0; ks < KCHUNK/32; ++ks){
      const us* pp = &sP[l16*KCHUNK + ks*32 + hf*8];      // LDS A operand
      v16bf a = load2(pp, pp + 16);
      #pragma unroll
      for (int nt = 0; nt < 4; ++nt){
        int c0 = wave*64 + nt*16;
        const us* vrow = vbase + (size_t)(c0 + l16)*SS + kc*KCHUNK + ks*32 + hf*8;
        O[nt] = bfwmma(a, load2(vrow, vrow + 16), O[nt]);
      }
    }
    __syncthreads();
  }

  // --- finalize: divide by l, store (b, s, h, c) bf16 ---
  #pragma unroll
  for (int nt = 0; nt < 4; ++nt){
    int cg = wave*64 + nt*16 + l16;
    #pragma unroll
    for (int r = 0; r < 8; ++r){
      int s = qt*16 + r + hf*8;
      float o = O[nt][r] / sL[r + hf*8];
      attn[(((size_t)b*SS + s)*HH + h)*CC + cg] = f2bf(o);
    }
  }
}

// ---------------- stage 3: output projection ----------------
// one wave -> 16x64 tile, K=2048
__global__ __launch_bounds__(32) void proj_gemm(
    const us* __restrict__ attn, const us* __restrict__ Wot,
    const float* __restrict__ bo, float* __restrict__ out)
{
  int lane = threadIdx.x;
  int hf = lane >> 4, l16 = lane & 15;
  int nBase = blockIdx.x * 64;
  int mBase = blockIdx.y * 16;
  const us* arow = attn + (size_t)(mBase + l16) * HC;
  v8f acc[4] = {};
  #pragma unroll 8
  for (int cs = 0; cs < HC/32; ++cs){
    v16bf a = load2(arow + cs*32 + hf*8, arow + cs*32 + 16 + hf*8);
    #pragma unroll
    for (int t = 0; t < 4; ++t){
      const us* bp = Wot + (size_t)(nBase + t*16 + l16) * HC + cs*32 + hf*8;
      acc[t] = bfwmma(a, load2(bp, bp + 16), acc[t]);
    }
  }
  #pragma unroll
  for (int t = 0; t < 4; ++t){
    int n = nBase + t*16 + l16;
    float bias = bo[n];
    #pragma unroll
    for (int r = 0; r < 8; ++r)
      out[(size_t)(mBase + r + hf*8)*CC + n] = acc[t][r] + bias;
  }
}

extern "C" void kernel_launch(void* const* d_in, const int* in_sizes, int n_in,
                              void* d_out, int out_size, void* d_ws, size_t ws_size,
                              hipStream_t stream)
{
  const float* x  = (const float*)d_in[0];
  const float* Wq = (const float*)d_in[1];
  const float* bq = (const float*)d_in[2];
  const float* Wk = (const float*)d_in[3];
  const float* bk = (const float*)d_in[4];
  const float* Wv = (const float*)d_in[5];
  const float* bv = (const float*)d_in[6];
  const float* Wo = (const float*)d_in[7];
  const float* bo = (const float*)d_in[8];
  float* out = (float*)d_out;

  char* ws = (char*)d_ws;
  size_t off = 0;
  us* xbf = (us*)(ws + off); off += (size_t)MM*CC*2;
  us* Wqt = (us*)(ws + off); off += (size_t)HC*CC*2;
  us* Wkt = (us*)(ws + off); off += (size_t)HC*CC*2;
  us* Wvt = (us*)(ws + off); off += (size_t)HC*CC*2;
  us* Wot = (us*)(ws + off); off += (size_t)CC*HC*2;
  us* qb  = (us*)(ws + off); off += (size_t)MM*HC*2;
  us* kb  = (us*)(ws + off); off += (size_t)MM*HC*2;
  us* vtb = (us*)(ws + off); off += (size_t)MM*HC*2;
  us* atn = (us*)(ws + off); off += (size_t)MM*HC*2;

  const float scale = 0.25f;  // 256^(-1/4), applied to both q and k weight/bias

  cvt_x<<<(MM*CC + 255)/256, 256, 0, stream>>>(x, xbf, MM*CC);
  int wtot = CC*HC;
  cvt_wT<<<(wtot + 255)/256, 256, 0, stream>>>(Wq, Wqt, CC, HC, scale);
  cvt_wT<<<(wtot + 255)/256, 256, 0, stream>>>(Wk, Wkt, CC, HC, scale);
  cvt_wT<<<(wtot + 255)/256, 256, 0, stream>>>(Wv, Wvt, CC, HC, 1.0f);
  cvt_wT<<<(wtot + 255)/256, 256, 0, stream>>>(Wo, Wot, HC, CC, 1.0f);

  qkv_gemm<<<dim3(HC/64, MM/16, 3), 32, 0, stream>>>(
      xbf, Wqt, Wkt, Wvt, bq, bk, bv, qb, kb, vtb);

  attn_kernel<<<BB*HH*(SS/16), 128, 0, stream>>>(qb, kb, vtb, atn);

  proj_gemm<<<dim3(CC/64, MM/16), 32, 0, stream>>>(atn, Wot, bo, out);
}
